// Attention_58695023067405
// MI455X (gfx1250) — matrix operand order repaired and checked by reading, not compiled, not run on hardware
//
#include <hip/hip_runtime.h>
#include <hip/hip_bf16.h>
#include <stdint.h>

// Problem constants (from reference): p,q are [B, L, D] float32.
#define B_  64
#define L_  512
#define D_  768
#define BL_ (B_ * L_)

typedef __attribute__((ext_vector_type(16))) __bf16       v16bf;
typedef __attribute__((ext_vector_type(8)))  __bf16       v8bf;
typedef __attribute__((ext_vector_type(8)))  float        v8f;
typedef __attribute__((ext_vector_type(4)))  unsigned int v4u;
typedef __attribute__((ext_vector_type(8)))  int          v8i_;
typedef __attribute__((ext_vector_type(4)))  int          v4i_;

__device__ __forceinline__ v8f wmma_bf16(v16bf a, v16bf b, v8f c) {
    // emits v_wmma_f32_16x16x32_bf16 (D = A[16x32] * B[32x16] + C, f32 acc)
    return __builtin_amdgcn_wmma_f32_16x16x32_bf16(
        /*neg_a=*/false, a, /*neg_b=*/false, b,
        /*c_mod=*/(short)0, c, /*reuse_a=*/false, /*reuse_b=*/false);
}

// ---------------------------------------------------------------------------
// TDM: DMA a 32(row) x 128(col) f32 tile (row stride 768 floats in memory)
// into LDS, with LDS padding of 2 DWORDs every 128 DWORDs so rows land at a
// stride of 130 floats (=> conflict-free column gathers later).
// D# packing per CDNA5 ISA ch.8.3/8.4. This toolchain's builtin takes 6 args:
// (uint32x4 g0, int32x8 g1, int32x4 g2, int32x4 g3, int32x8, i32 cpol).
// ---------------------------------------------------------------------------
__device__ __forceinline__ void tdm_load_tile(uint32_t lds_off, const float* gptr) {
    const uint64_t ga = (uint64_t)(uintptr_t)gptr;
    v4u g0;
    g0[0] = 1u;                                   // count=1, user descriptor
    g0[1] = lds_off;                              // lds_addr (bytes)
    g0[2] = (uint32_t)ga;                         // global_addr[31:0]
    g0[3] = (uint32_t)(ga >> 32) | (2u << 30);    // global_addr[56:32] | type=2
    v8i_ g1;
    g1[0] = (2 << 16)      // data_size = 4 bytes
          | (1 << 20)      // pad_enable
          | (6 << 22)      // pad_interval: every 128 DWORDs
          | (1 << 25);     // pad_amount: 2 DWORDs
    g1[1] = (int)(768u << 16);   // tensor_dim0 = 768 (never OOB for our tiles)
    g1[2] = (int)(512u << 16);   // tensor_dim1 = 512
    g1[3] = (int)(128u << 16);   // tile_dim0  = 128
    g1[4] = 32;                  // tile_dim1  = 32, tile_dim2 = 0
    g1[5] = 768;                 // tensor_dim0_stride[31:0] = 768
    g1[6] = 0;                   // stride hi / dim1_stride lo
    g1[7] = 0;
    v4i_ z4 = {};
    v8i_ z8 = {};
    __builtin_amdgcn_tensor_load_to_lds(g0, g1, z4, z4, z8, 0);
}

// ---------------------------------------------------------------------------
// Kernel 1: reciprocal L2 norms of every row of p and q (wave32 per row).
// ---------------------------------------------------------------------------
__global__ void __launch_bounds__(256)
norms_kernel(const float* __restrict__ p, const float* __restrict__ q,
             float* __restrict__ rp, float* __restrict__ rq) {
    const int wave = (blockIdx.x * 256 + threadIdx.x) >> 5;   // [0, 2*BL)
    const int lane = threadIdx.x & 31;
    const bool isQ = wave >= BL_;
    const int  row = isQ ? (wave - BL_) : wave;
    const float* r = (isQ ? q : p) + (size_t)row * D_;

    float s = 0.f;
    #pragma unroll
    for (int d = 0; d < D_ / 32; ++d) {
        float v = r[lane + d * 32];
        s += v * v;
    }
    #pragma unroll
    for (int off = 16; off > 0; off >>= 1) s += __shfl_xor(s, off, 32);
    if (lane == 0) (isQ ? rq : rp)[row] = 1.0f / fmaxf(sqrtf(s), 1e-8f);
}

// ---------------------------------------------------------------------------
// Kernel 2: ea[b,i,j] = exp( (p_i . q_j) * rp_i * rq_j )  stored as bf16.
// bf16 WMMA GEMM p[512x768] x q^T[768x512], 128x128 block tile, 8 waves,
// each wave owns a 32x64 tile (2 M-tiles x 4 N-tiles of 16x16).
// ---------------------------------------------------------------------------
__global__ void __launch_bounds__(256)
att_exp_kernel(const float* __restrict__ p, const float* __restrict__ q,
               const float* __restrict__ rp, const float* __restrict__ rq,
               __bf16* __restrict__ ea) {
    const int bid  = blockIdx.x;
    const int b    = bid >> 4;                  // 4x4 = 16 tiles per batch
    const int tile = bid & 15;
    const int bRow = (tile >> 2) * 128;
    const int bCol = (tile & 3) * 128;

    const int wid  = threadIdx.x >> 5;
    const int lane = threadIdx.x & 31;
    const int llo  = lane & 15;
    const int lhi  = lane >> 4;
    const int wRow = bRow + (wid & 3) * 32;     // 4 waves along M
    const int wCol = bCol + (wid >> 2) * 64;    // 2 waves along N

    const float* pb = p + (size_t)b * L_ * D_;
    const float* qb = q + (size_t)b * L_ * D_;

    v8f acc[2][4];
    #pragma unroll
    for (int tm = 0; tm < 2; ++tm)
        #pragma unroll
        for (int u = 0; u < 4; ++u) acc[tm][u] = v8f{};

    for (int kb = 0; kb < D_; kb += 32) {
        // A fragments: lane holds row (wRow+tm*16+llo); elems 0-15 at
        // K = kb + lhi*16 + (0..15) (contiguous, matches B).
        v16bf afrag[2];
        #pragma unroll
        for (int tm = 0; tm < 2; ++tm) {
            const float* ar = pb + (size_t)(wRow + tm * 16 + llo) * D_ + kb + lhi * 16;
            float4 f0 = *(const float4*)(ar);
            float4 f1 = *(const float4*)(ar + 4);
            float4 f2 = *(const float4*)(ar + 8);
            float4 f3 = *(const float4*)(ar + 12);
            v16bf a;
            a[0]=(__bf16)f0.x; a[1]=(__bf16)f0.y; a[2]=(__bf16)f0.z; a[3]=(__bf16)f0.w;
            a[4]=(__bf16)f1.x; a[5]=(__bf16)f1.y; a[6]=(__bf16)f1.z; a[7]=(__bf16)f1.w;
            a[8]=(__bf16)f2.x; a[9]=(__bf16)f2.y; a[10]=(__bf16)f2.z; a[11]=(__bf16)f2.w;
            a[12]=(__bf16)f3.x; a[13]=(__bf16)f3.y; a[14]=(__bf16)f3.z; a[15]=(__bf16)f3.w;
            afrag[tm] = a;
        }
        // B fragments: B[k][n] = q[row n][k]; lane holds col n = llo,
        // K = kb + lhi*16 + (0..15) => 16 contiguous floats of a q row.
        v16bf bfrag[4];
        #pragma unroll
        for (int u = 0; u < 4; ++u) {
            const float* br = qb + (size_t)(wCol + u * 16 + llo) * D_ + kb + lhi * 16;
            float4 f0 = *(const float4*)(br);
            float4 f1 = *(const float4*)(br + 4);
            float4 f2 = *(const float4*)(br + 8);
            float4 f3 = *(const float4*)(br + 12);
            v16bf v;
            v[0]=(__bf16)f0.x; v[1]=(__bf16)f0.y; v[2]=(__bf16)f0.z; v[3]=(__bf16)f0.w;
            v[4]=(__bf16)f1.x; v[5]=(__bf16)f1.y; v[6]=(__bf16)f1.z; v[7]=(__bf16)f1.w;
            v[8]=(__bf16)f2.x; v[9]=(__bf16)f2.y; v[10]=(__bf16)f2.z; v[11]=(__bf16)f2.w;
            v[12]=(__bf16)f3.x; v[13]=(__bf16)f3.y; v[14]=(__bf16)f3.z; v[15]=(__bf16)f3.w;
            bfrag[u] = v;
        }
        #pragma unroll
        for (int tm = 0; tm < 2; ++tm)
            #pragma unroll
            for (int u = 0; u < 4; ++u)
                acc[tm][u] = wmma_bf16(afrag[tm], bfrag[u], acc[tm][u]);
    }

    // Epilogue: scale by rp_i * rq_j, exp, store bf16.
    float rpv[2][8];
    #pragma unroll
    for (int tm = 0; tm < 2; ++tm)
        #pragma unroll
        for (int v = 0; v < 8; ++v)
            rpv[tm][v] = rp[b * L_ + wRow + tm * 16 + v + 8 * lhi];
    float rqv[4];
    #pragma unroll
    for (int u = 0; u < 4; ++u) rqv[u] = rq[b * L_ + wCol + u * 16 + llo];

    __bf16* eb = ea + (size_t)b * L_ * L_;
    #pragma unroll
    for (int tm = 0; tm < 2; ++tm)
        #pragma unroll
        for (int u = 0; u < 4; ++u)
            #pragma unroll
            for (int v = 0; v < 8; ++v) {
                const int i = wRow + tm * 16 + v + 8 * lhi;   // C/D: M = v + 8*(lane/16)
                const int j = wCol + u * 16 + llo;            // C/D: N = lane%16
                float val = acc[tm][u][v] * rpv[tm][v] * rqv[u];
                eb[(size_t)i * L_ + j] = (__bf16)__expf(val); // cos in [-1,1]: no max shift needed
            }
}

// ---------------------------------------------------------------------------
// Kernel 3a: reciprocal row sums of exp(att)  (softmax over j denominator).
// ---------------------------------------------------------------------------
__global__ void __launch_bounds__(256)
rowsum_kernel(const __bf16* __restrict__ ea, float* __restrict__ rrs) {
    const int wave = (blockIdx.x * 256 + threadIdx.x) >> 5;   // = b*L + i
    const int lane = threadIdx.x & 31;
    const __bf16* r = ea + (size_t)wave * L_;
    float s = 0.f;
    #pragma unroll
    for (int k = 0; k < L_ / 32; ++k) s += (float)r[lane + k * 32];
    #pragma unroll
    for (int off = 16; off > 0; off >>= 1) s += __shfl_xor(s, off, 32);
    if (lane == 0) rrs[wave] = 1.0f / s;
}

// ---------------------------------------------------------------------------
// Kernel 3b: reciprocal column sums of exp(att) (softmax over i denominator).
// ---------------------------------------------------------------------------
__global__ void __launch_bounds__(256)
colsum_kernel(const __bf16* __restrict__ ea, float* __restrict__ rcs) {
    const int tid = blockIdx.x * 256 + threadIdx.x;           // = b*L + j
    const int b = tid / L_;
    const int j = tid - b * L_;
    const __bf16* c = ea + (size_t)b * L_ * L_ + j;
    float s = 0.f;
    for (int i = 0; i < L_; ++i) s += (float)c[(size_t)i * L_];
    rcs[tid] = 1.0f / s;
}

// ---------------------------------------------------------------------------
// Kernel 4: fused output GEMMs.
//   out_p[i,d] = (1/rowsum_i) * sum_j exp(att[i,j]) * q[j,d]
//   out_q[i,d] =               sum_j exp(att[i,j]) * (p[j,d] * rcs_j)
// A = exp(att) bf16 (shared between both). B tiles (32j x 128d f32) are
// DMA'd into LDS by the Tensor Data Mover (wave 0 issues, s_wait_tensorcnt,
// block barrier). bf16 convert + rcs_j scale fused into fragment build.
// Block tile 128(M) x 128(N), K-step 32; wave tile 32x64.
// ---------------------------------------------------------------------------
__global__ void __launch_bounds__(256)
out_kernel(const float* __restrict__ p, const float* __restrict__ q,
           const __bf16* __restrict__ ea, const float* __restrict__ rrs,
           const float* __restrict__ rcs,
           float* __restrict__ outp, float* __restrict__ outq) {
    const int bid  = blockIdx.x;
    const int b    = bid / 24;                   // 4 (M) x 6 (N) tiles per batch
    const int tile = bid - b * 24;
    const int bRow = (tile / 6) * 128;           // i
    const int bCol = (tile % 6) * 128;           // d

    // TDM pads 2 DWORDs per 128 -> effective row stride 130 floats.
    __shared__ float qsf[32][130];
    __shared__ float psf[32][130];

    const int wid  = threadIdx.x >> 5;
    const int lane = threadIdx.x & 31;
    const int llo  = lane & 15;
    const int lhi  = lane >> 4;
    const int wRow    = bRow + (wid & 3) * 32;
    const int wColLoc = (wid >> 2) * 64;         // N offset within block tile

    const float*  pb = p + (size_t)b * L_ * D_;
    const float*  qb = q + (size_t)b * L_ * D_;
    const __bf16* eb = ea + (size_t)b * L_ * L_;

    // Low 32 bits of a generic LDS pointer are the LDS byte offset.
    const uint32_t ldsQ = (uint32_t)(uintptr_t)&qsf[0][0];
    const uint32_t ldsP = (uint32_t)(uintptr_t)&psf[0][0];

    v8f accP[2][4], accQ[2][4];
    #pragma unroll
    for (int tm = 0; tm < 2; ++tm)
        #pragma unroll
        for (int u = 0; u < 4; ++u) { accP[tm][u] = v8f{}; accQ[tm][u] = v8f{}; }

    for (int kb = 0; kb < L_; kb += 32) {
        __syncthreads();                         // previous LDS consumers done
        if (wid == 0) {                          // one TDM issue per block
            tdm_load_tile(ldsQ, qb + (size_t)kb * D_ + bCol);
            tdm_load_tile(ldsP, pb + (size_t)kb * D_ + bCol);
            __builtin_amdgcn_s_wait_tensorcnt(0);
        }
        __syncthreads();

        // 1/colsum for this K block: j = kb + lhi*16 + e.
        const float* rcb = rcs + b * L_ + kb + lhi * 16;
        float4 rc0 = *(const float4*)(rcb);
        float4 rc1 = *(const float4*)(rcb + 4);
        float4 rc2 = *(const float4*)(rcb + 8);
        float4 rc3 = *(const float4*)(rcb + 12);
        const float rcv[16] = {rc0.x, rc0.y, rc0.z, rc0.w,
                               rc1.x, rc1.y, rc1.z, rc1.w,
                               rc2.x, rc2.y, rc2.z, rc2.w,
                               rc3.x, rc3.y, rc3.z, rc3.w};

        // A fragments straight from bf16 exp(att), K = kb + lhi*16 + e (contiguous, same as B).
        v16bf afrag[2];
        #pragma unroll
        for (int tm = 0; tm < 2; ++tm) {
            const __bf16* ar = eb + (size_t)(wRow + tm * 16 + llo) * L_ + kb + lhi * 16;
            v8bf lo = *(const v8bf*)(ar);        // K = kb + lhi*16 + (0..7)
            v8bf hi = *(const v8bf*)(ar + 8);    // K = kb + lhi*16 + 8 + (0..7)
            v16bf a;
            #pragma unroll
            for (int e = 0; e < 8; ++e) { a[e] = lo[e]; a[8 + e] = hi[e]; }
            afrag[tm] = a;
        }

        // B fragments: column gather from f32 LDS tile (stride 130 =>
        // 32 lanes hit 32 distinct banks), fused bf16 convert (+ rcs scale).
        #pragma unroll
        for (int u = 0; u < 4; ++u) {
            const int nloc = wColLoc + u * 16 + llo;
            const float* bq = &qsf[lhi * 16][nloc];
            const float* bp = &psf[lhi * 16][nloc];
            v16bf bQ, bP;
            #pragma unroll
            for (int e = 0; e < 16; ++e) {
                bQ[e] = (__bf16)bq[e * 130];
                bP[e] = (__bf16)(bp[e * 130] * rcv[e]);
            }
            #pragma unroll
            for (int tm = 0; tm < 2; ++tm) {
                accP[tm][u] = wmma_bf16(afrag[tm], bQ, accP[tm][u]);
                accQ[tm][u] = wmma_bf16(afrag[tm], bP, accQ[tm][u]);
            }
        }
    }

    // Epilogue: out_p rows scaled by 1/rowsum; out_q already fully scaled.
    #pragma unroll
    for (int tm = 0; tm < 2; ++tm) {
        float rr[8];
        #pragma unroll
        for (int v = 0; v < 8; ++v)
            rr[v] = rrs[b * L_ + wRow + tm * 16 + v + 8 * lhi];
        #pragma unroll
        for (int u = 0; u < 4; ++u)
            #pragma unroll
            for (int v = 0; v < 8; ++v) {
                const int i = wRow + tm * 16 + v + 8 * lhi;
                const int d = bCol + wColLoc + u * 16 + llo;
                const size_t o = ((size_t)b * L_ + i) * D_ + d;
                outp[o] = accP[tm][u][v] * rr[v];
                outq[o] = accQ[tm][u][v];
            }
    }
}

// ---------------------------------------------------------------------------
extern "C" void kernel_launch(void* const* d_in, const int* in_sizes, int n_in,
                              void* d_out, int out_size, void* d_ws, size_t ws_size,
                              hipStream_t stream) {
    const float* p = (const float*)d_in[0];
    const float* q = (const float*)d_in[1];
    float* out = (float*)d_out;

    // Workspace layout (~34 MB): 4 reciprocal-vector arrays + bf16 exp(att).
    float*  rp  = (float*)d_ws;          // [B*L]
    float*  rq  = rp + BL_;              // [B*L]
    float*  rrs = rq + BL_;              // [B*L]
    float*  rcs = rrs + BL_;             // [B*L]
    __bf16* ea  = (__bf16*)(rcs + BL_);  // [B*L*L]

    norms_kernel  <<<(2 * BL_) / 8, 256, 0, stream>>>(p, q, rp, rq);
    att_exp_kernel<<<B_ * 16,       256, 0, stream>>>(p, q, rp, rq, ea);
    rowsum_kernel <<<BL_ / 8,       256, 0, stream>>>(ea, rrs);
    colsum_kernel <<<BL_ / 256,     256, 0, stream>>>(ea, rcs);
    out_kernel    <<<B_ * 24,       256, 0, stream>>>(p, q, ea, rrs, rcs,
                                                      out, out + (size_t)B_ * L_ * D_);
}
